// AutoregressiveVAE_54984171323964
// MI455X (gfx1250) — matrix operand order repaired and checked by reading, not compiled, not run on hardware
//
#include <hip/hip_runtime.h>
#include <hip/hip_bf16.h>

#define DEVI __device__ __forceinline__

typedef __attribute__((ext_vector_type(16))) __bf16 v16bf;
typedef __attribute__((ext_vector_type(8)))  __bf16 v8bf;
typedef __attribute__((ext_vector_type(8)))  float  v8f;

namespace {

constexpr int Bn = 64;    // batch
constexpr int Tn = 128;   // timesteps
constexpr int Dn = 256;   // input/latent dim
constexpr int Hn = 1024;  // hidden dim
constexpr int FH = 4096;  // 4*H (gates)

DEVI float sigm(float x) { return 1.0f / (1.0f + __expf(-x)); }

DEVI float tanh_fast(float x) {
  x = fminf(fmaxf(x, -30.0f), 30.0f);
  float e = __expf(-2.0f * x);
  return (1.0f - e) / (1.0f + e);
}

DEVI v8f vzero() {
  v8f v;
#pragma unroll
  for (int i = 0; i < 8; ++i) v[i] = 0.0f;
  return v;
}

// A-fragment: 16x32 bf16 tile of activations X (row-major, leading dim ld).
// Contiguous k order: lanes 0-15 hold row m, K chunks {kb+0..7, kb+8..15};
//             lanes 16-31 hold row m, K chunks {kb+16..23, kb+24..31}.
DEVI v16bf load_a_frag(const __bf16* __restrict__ X, int ld, int m_base, int kb, int lane) {
  const int row = m_base + (lane & 15);
  const int k0  = kb + ((lane & 16) ? 16 : 0);
  const __bf16* p = X + (size_t)row * ld + k0;
  v8bf lo = *reinterpret_cast<const v8bf*>(p);        // 16B contiguous
  v8bf hi = *reinterpret_cast<const v8bf*>(p + 8);    // 16B contiguous
  v16bf a;
#pragma unroll
  for (int i = 0; i < 8; ++i) { a[i] = lo[i]; a[i + 8] = hi[i]; }
  return a;
}

// B-fragment straight from global (used by the small projection heads):
// B[k][n] = W[n][k], W row-major [N, ld].
DEVI v16bf load_b_frag(const __bf16* __restrict__ W, int ld, int n_base, int kb, int lane) {
  const int n  = n_base + (lane & 15);
  const int k0 = kb + ((lane & 16) ? 16 : 0);
  return *reinterpret_cast<const v16bf*>(W + (size_t)n * ld + k0);
}

DEVI v8f wmma_bf16(v16bf a, v16bf b, v8f c) {
  // (neg_a, A, neg_b, B, c_mod, C, reuse_a, reuse_b)
  return __builtin_amdgcn_wmma_f32_16x16x32_bf16(false, a, false, b, (short)0, c, false, false);
}

// Async copy of 16 bytes global -> LDS (VDST = LDS byte offset VGPR,
// VADDR = 64-bit global address).  Tracked by ASYNCcnt.
DEVI void async_g2l_b128(uint32_t lds_off, const __bf16* gsrc) {
  asm volatile("global_load_async_to_lds_b128 %0, %1, off"
               :: "v"(lds_off), "v"(gsrc) : "memory");
}

DEVI void wait_asynccnt0() {
  asm volatile("s_wait_asynccnt 0" ::: "memory");
}

} // namespace

// ---------------------------------------------------------------------------
// Fused LSTM layer step: gates = X @ Wih^T + Hprev @ Whh^T + b, then cell
// update.  One block = one 16-wide hidden column tile (j-tile) incl. all four
// gates; its 4 waves are the 4 M-tiles of the batch, so the gate weight tiles
// are staged ONCE per block into LDS (async, double buffered) and consumed by
// all 4 waves -> 4x less L2 weight traffic than per-wave global loads.
// Stage = 128 k-elements: 4 gates x 16 rows x 256B = 16KB, 2 buffers = 32KB.
// Grid: 64 blocks (j-tiles) x 128 threads (4 waves = M-tiles).
// ---------------------------------------------------------------------------
__global__ void __launch_bounds__(128)
lstm_layer_kernel(const __bf16* __restrict__ X, int Kx,
                  const __bf16* __restrict__ Hprev,   // [B, H] bf16
                  const __bf16* __restrict__ Wih,     // [4H, Kx] bf16
                  const __bf16* __restrict__ Whh,     // [4H, H] bf16
                  const float* __restrict__ bias,     // [4H] fp32
                  float* __restrict__ Cst,            // [B, H] fp32 in/out
                  __bf16* __restrict__ Hnew) {        // [B, H] bf16 out
  __shared__ __align__(128) __bf16 smem[2 * 8192];    // 2 x 16KB stages

  const int tid    = threadIdx.x;
  const int lane   = tid & 31;
  const int m_base = (tid >> 5) << 4;     // wave id -> batch row tile
  const int nj     = blockIdx.x << 4;     // hidden column tile

  const uint32_t lds_base = (uint32_t)(uintptr_t)smem;  // LDS byte offset
  const int Sx = Kx >> 7;                 // stages sourced from Wih
  const int S  = (Kx + Hn) >> 7;          // total stages (concatenated K)

  // Cooperative async stage: 128 threads x 2 units x 4 k-steps, 16B each.
  auto issue_stage = [&](int s) {
    const __bf16* W;
    int ld, kb0;
    if (s < Sx) { W = Wih; ld = Kx; kb0 = s << 7; }
    else        { W = Whh; ld = Hn; kb0 = (s - Sx) << 7; }
    const uint32_t dst0 = lds_base + (uint32_t)((s & 1) * 16384);
#pragma unroll
    for (int ks = 0; ks < 4; ++ks) {
#pragma unroll
      for (int half = 0; half < 2; ++half) {
        const int u   = tid * 2 + half;   // 0..255 units of 16B per k-step
        const int g   = u >> 6;           // gate
        const int row = (u >> 2) & 15;    // row within gate tile
        const int seg = u & 3;            // 16B segment within 64B row
        const __bf16* src =
            W + (size_t)(g * Hn + nj + row) * ld + kb0 + ks * 32 + seg * 8;
        const uint32_t dst =
            dst0 + (uint32_t)(ks * 4096 + (g * 16 + row) * 64 + seg * 16);
        async_g2l_b128(dst, src);
      }
    }
  };

  v8f acc[4] = {vzero(), vzero(), vzero(), vzero()};

  issue_stage(0);
  wait_asynccnt0();
  __syncthreads();

  for (int s = 0; s < S; ++s) {
    if (s + 1 < S) issue_stage(s + 1);    // overlap copy with compute

    const __bf16* bufp = smem + (size_t)((s & 1) * 8192);
#pragma unroll
    for (int ks = 0; ks < 4; ++ks) {
      const int kc = (s << 7) + ks * 32;  // concatenated-K position
      v16bf a = (kc < Kx) ? load_a_frag(X, Kx, m_base, kc, lane)
                          : load_a_frag(Hprev, Hn, m_base, kc - Kx, lane);
#pragma unroll
      for (int g = 0; g < 4; ++g) {
        // B fragment from LDS: lane < 16 -> row n, k 0..15; lane >= 16 ->
        // row n, k 16..31 (32 contiguous bytes of the staged 64B row).
        const __bf16* bp = bufp + ks * 2048 + (g * 16 + (lane & 15)) * 32 +
                           ((lane & 16) ? 16 : 0);
        v16bf b = *reinterpret_cast<const v16bf*>(bp);
        acc[g] = wmma_bf16(a, b, acc[g]);
      }
    }

    wait_asynccnt0();                     // next stage landed in other buffer
    __syncthreads();                      // ...and everyone is done reading
  }

  // C/D layout: element r of v8f at lane -> row m_base + r (+8 for lanes>=16),
  // col = nj + (lane & 15).
  const int j  = nj + (lane & 15);
  const int mo = (lane & 16) ? 8 : 0;
  const float bi  = bias[j];
  const float bf_ = bias[Hn + j];
  const float bg  = bias[2 * Hn + j];
  const float bo  = bias[3 * Hn + j];
#pragma unroll
  for (int r = 0; r < 8; ++r) {
    const int m = m_base + mo + r;
    const size_t o = (size_t)m * Hn + j;
    const float ig = sigm(acc[0][r] + bi);
    const float fg = sigm(acc[1][r] + bf_);
    const float gg = tanh_fast(acc[2][r] + bg);
    const float og = sigm(acc[3][r] + bo);
    const float cn = fg * Cst[o] + ig * gg;
    Cst[o] = cn;
    Hnew[o] = (__bf16)(og * tanh_fast(cn));
  }
}

// ---------------------------------------------------------------------------
// Reparameterization head: mu/logvar = clip(Henc @ W^T + b), z = mu + eps*std.
// Grid: 16 blocks x 128 threads = 64 waves = 4 M-tiles x 16 N-tiles (D=256).
// ---------------------------------------------------------------------------
__global__ void __launch_bounds__(128)
proj_enc_kernel(const __bf16* __restrict__ Henc,
                const __bf16* __restrict__ Wmu, const __bf16* __restrict__ Wlv,
                const float* __restrict__ bmu, const float* __restrict__ blv,
                const float* __restrict__ eps, int t,
                float* __restrict__ out_mu, float* __restrict__ out_lv,
                float* __restrict__ out_z, __bf16* __restrict__ Zbf) {
  const int wave   = (blockIdx.x * blockDim.x + threadIdx.x) >> 5;
  const int lane   = threadIdx.x & 31;
  const int m_base = (wave >> 4) << 4;
  const int nb     = (wave & 15) << 4;

  v8f a0 = vzero(), a1 = vzero();
  for (int kb = 0; kb < Hn; kb += 32) {
    v16bf a = load_a_frag(Henc, Hn, m_base, kb, lane);
    a0 = wmma_bf16(a, load_b_frag(Wmu, Hn, nb, kb, lane), a0);
    a1 = wmma_bf16(a, load_b_frag(Wlv, Hn, nb, kb, lane), a1);
  }

  const int j  = nb + (lane & 15);
  const int mo = (lane & 16) ? 8 : 0;
  const float bm = bmu[j];
  const float bl = blv[j];
#pragma unroll
  for (int r = 0; r < 8; ++r) {
    const int m = m_base + mo + r;
    float mu = fminf(fmaxf(a0[r] + bm, -20.0f), 20.0f);
    float lv = fminf(fmaxf(a1[r] + bl, -20.0f), 20.0f);
    float sd = __expf(0.5f * lv);
    const size_t o = ((size_t)m * Tn + t) * Dn + j;
    float z = mu + eps[o] * sd;
    out_mu[o] = mu;
    out_lv[o] = lv;
    out_z[o]  = z;
    Zbf[(size_t)m * Dn + j] = (__bf16)z;
  }
}

// ---------------------------------------------------------------------------
// Output head: logits = Hdec @ Wout^T + b, rec = sigmoid(logits); logits are
// also kept as c_prev for the next step's autoregressive encoder input.
// ---------------------------------------------------------------------------
__global__ void __launch_bounds__(128)
proj_out_kernel(const __bf16* __restrict__ Hdec, const __bf16* __restrict__ Wout,
                const float* __restrict__ bout, int t,
                float* __restrict__ out_rec, float* __restrict__ out_logits,
                float* __restrict__ lgp) {
  const int wave   = (blockIdx.x * blockDim.x + threadIdx.x) >> 5;
  const int lane   = threadIdx.x & 31;
  const int m_base = (wave >> 4) << 4;
  const int nb     = (wave & 15) << 4;

  v8f a0 = vzero();
  for (int kb = 0; kb < Hn; kb += 32) {
    v16bf a = load_a_frag(Hdec, Hn, m_base, kb, lane);
    a0 = wmma_bf16(a, load_b_frag(Wout, Hn, nb, kb, lane), a0);
  }

  const int j  = nb + (lane & 15);
  const int mo = (lane & 16) ? 8 : 0;
  const float bb = bout[j];
#pragma unroll
  for (int r = 0; r < 8; ++r) {
    const int m = m_base + mo + r;
    float lg = a0[r] + bb;
    const size_t o = ((size_t)m * Tn + t) * Dn + j;
    out_logits[o] = lg;
    out_rec[o]    = sigm(lg);
    lgp[(size_t)m * Dn + j] = lg;
  }
}

// enc_in = [x_t | x_t - sigmoid(logits_prev)] in bf16.  16384 threads.
__global__ void prep_enc_in_kernel(const float* __restrict__ x,
                                   const float* __restrict__ lgp, int t,
                                   __bf16* __restrict__ enc_in) {
  const int idx = blockIdx.x * blockDim.x + threadIdx.x;  // Bn*Dn
  const int b = idx >> 8;
  const int d = idx & 255;
  const float xv = x[((size_t)b * Tn + t) * Dn + d];
  const float xh = xv - sigm(lgp[(size_t)b * Dn + d]);
  enc_in[(size_t)b * (2 * Dn) + d]      = (__bf16)xv;
  enc_in[(size_t)b * (2 * Dn) + Dn + d] = (__bf16)xh;
}

// One-time fp32 -> bf16 weight conversion.
__global__ void f2bf_kernel(const float* __restrict__ s, __bf16* __restrict__ d, int n) {
  const int i = blockIdx.x * blockDim.x + threadIdx.x;
  if (i < n) d[i] = (__bf16)s[i];
}

// ---------------------------------------------------------------------------
extern "C" void kernel_launch(void* const* d_in, const int* in_sizes, int n_in,
                              void* d_out, int out_size, void* d_ws, size_t ws_size,
                              hipStream_t stream) {
  (void)in_sizes; (void)n_in; (void)out_size; (void)ws_size;

  const float* x        = (const float*)d_in[0];
  const float* eps      = (const float*)d_in[1];
  const float* enc_Wih0 = (const float*)d_in[2];
  const float* enc_Wih  = (const float*)d_in[3];
  const float* enc_Whh  = (const float*)d_in[4];
  const float* enc_b    = (const float*)d_in[5];
  const float* W_mu     = (const float*)d_in[6];
  const float* b_mu     = (const float*)d_in[7];
  const float* W_lv     = (const float*)d_in[8];
  const float* b_lv     = (const float*)d_in[9];
  const float* dec_Wih0 = (const float*)d_in[10];
  const float* dec_Wih  = (const float*)d_in[11];
  const float* dec_Whh  = (const float*)d_in[12];
  const float* dec_b    = (const float*)d_in[13];
  const float* W_out    = (const float*)d_in[14];
  const float* b_out    = (const float*)d_in[15];

  float* out = (float*)d_out;
  const size_t N1 = (size_t)Bn * Tn * Dn;  // one [B,T,D] slab
  float* out_rec = out;
  float* out_log = out + N1;
  float* out_mu  = out + 2 * N1;
  float* out_lv  = out + 3 * N1;
  float* out_z   = out + 4 * N1;

  // ---- workspace carve (bf16 weight cache + state), ~95 MB total ----
  char* p = (char*)d_ws;
  auto take = [&](size_t bytes) -> char* {
    char* r = p;
    p += (bytes + 255) & ~(size_t)255;
    return r;
  };
  __bf16* w_enc_Wih0 = (__bf16*)take((size_t)FH * (2 * Dn) * 2);
  __bf16* w_enc_Wih  = (__bf16*)take((size_t)2 * FH * Hn * 2);
  __bf16* w_enc_Whh  = (__bf16*)take((size_t)3 * FH * Hn * 2);
  __bf16* w_dec_Wih0 = (__bf16*)take((size_t)FH * Dn * 2);
  __bf16* w_dec_Wih  = (__bf16*)take((size_t)2 * FH * Hn * 2);
  __bf16* w_dec_Whh  = (__bf16*)take((size_t)3 * FH * Hn * 2);
  __bf16* w_mu  = (__bf16*)take((size_t)Dn * Hn * 2);
  __bf16* w_lv  = (__bf16*)take((size_t)Dn * Hn * 2);
  __bf16* w_out = (__bf16*)take((size_t)Dn * Hn * 2);
  __bf16* hbuf  = (__bf16*)take((size_t)2 * 6 * Bn * Hn * 2);  // [parity][6][B,H]
  float*  cbuf  = (float*) take((size_t)6 * Bn * Hn * 4);       // [6][B,H]
  __bf16* enc_in = (__bf16*)take((size_t)Bn * (2 * Dn) * 2);
  __bf16* z_bf   = (__bf16*)take((size_t)Bn * Dn * 2);
  float*  lgp    = (float*) take((size_t)Bn * Dn * 4);

  auto Hb = [&](int parity, int idx) -> __bf16* {
    return hbuf + (size_t)(parity * 6 + idx) * Bn * Hn;
  };
  auto Cb = [&](int idx) -> float* { return cbuf + (size_t)idx * Bn * Hn; };

  // ---- one-time (per launch) weight conversion to resident bf16 cache ----
  auto conv = [&](const float* s, __bf16* d, size_t n) {
    f2bf_kernel<<<(int)((n + 255) / 256), 256, 0, stream>>>(s, d, (int)n);
  };
  conv(enc_Wih0, w_enc_Wih0, (size_t)FH * (2 * Dn));
  conv(enc_Wih,  w_enc_Wih,  (size_t)2 * FH * Hn);
  conv(enc_Whh,  w_enc_Whh,  (size_t)3 * FH * Hn);
  conv(dec_Wih0, w_dec_Wih0, (size_t)FH * Dn);
  conv(dec_Wih,  w_dec_Wih,  (size_t)2 * FH * Hn);
  conv(dec_Whh,  w_dec_Whh,  (size_t)3 * FH * Hn);
  conv(W_mu,  w_mu,  (size_t)Dn * Hn);
  conv(W_lv,  w_lv,  (size_t)Dn * Hn);
  conv(W_out, w_out, (size_t)Dn * Hn);

  // zero recurrent state (parity-0 h, all c, logits_prev)
  hipMemsetAsync(hbuf, 0, (size_t)6 * Bn * Hn * 2, stream);
  hipMemsetAsync(cbuf, 0, (size_t)6 * Bn * Hn * 4, stream);
  hipMemsetAsync(lgp,  0, (size_t)Bn * Dn * 4, stream);

  // ---- recurrent timestep loop ----
  for (int t = 0; t < Tn; ++t) {
    const int pa = t & 1, qa = pa ^ 1;

    prep_enc_in_kernel<<<64, 256, 0, stream>>>(x, lgp, t, enc_in);

    // encoder stack
    lstm_layer_kernel<<<64, 128, 0, stream>>>(enc_in, 2 * Dn, Hb(pa, 0),
        w_enc_Wih0, w_enc_Whh, enc_b, Cb(0), Hb(qa, 0));
    lstm_layer_kernel<<<64, 128, 0, stream>>>(Hb(qa, 0), Hn, Hb(pa, 1),
        w_enc_Wih, w_enc_Whh + (size_t)FH * Hn, enc_b + FH, Cb(1), Hb(qa, 1));
    lstm_layer_kernel<<<64, 128, 0, stream>>>(Hb(qa, 1), Hn, Hb(pa, 2),
        w_enc_Wih + (size_t)FH * Hn, w_enc_Whh + (size_t)2 * FH * Hn,
        enc_b + 2 * FH, Cb(2), Hb(qa, 2));

    proj_enc_kernel<<<16, 128, 0, stream>>>(Hb(qa, 2), w_mu, w_lv, b_mu, b_lv,
        eps, t, out_mu, out_lv, out_z, z_bf);

    // decoder stack
    lstm_layer_kernel<<<64, 128, 0, stream>>>(z_bf, Dn, Hb(pa, 3),
        w_dec_Wih0, w_dec_Whh, dec_b, Cb(3), Hb(qa, 3));
    lstm_layer_kernel<<<64, 128, 0, stream>>>(Hb(qa, 3), Hn, Hb(pa, 4),
        w_dec_Wih, w_dec_Whh + (size_t)FH * Hn, dec_b + FH, Cb(4), Hb(qa, 4));
    lstm_layer_kernel<<<64, 128, 0, stream>>>(Hb(qa, 4), Hn, Hb(pa, 5),
        w_dec_Wih + (size_t)FH * Hn, w_dec_Whh + (size_t)2 * FH * Hn,
        dec_b + 2 * FH, Cb(5), Hb(qa, 5));

    proj_out_kernel<<<16, 128, 0, stream>>>(Hb(qa, 5), w_out, b_out, t,
        out_rec, out_log, lgp);
  }
}
